// GCPNet_4509715661506
// MI455X (gfx1250) — matrix operand order repaired and checked by reading, not compiled, not run on hardware
//
#include <hip/hip_runtime.h>
#include <hip/hip_bf16.h>

// ---------------------------------------------------------------------------
// NeuroSAT-style recurrent GNN for MI455X (gfx1250).
// - Adjacency matrices converted once to bf16 (M_vv: 256MB fp32 -> 128MB bf16,
//   L2-resident in the 192MB L2); all GEMMs via v_wmma_f32_16x16x32_bf16.
// - K=8192 GEMMs use deterministic split-K (per-chunk fp32 partials + ordered
//   reduction) to raise wave parallelism on the serial critical path.
// - Split-K inner loop: manual register double-buffering (load k+32 fragments
//   before the k WMMAs) so WMMAs wait on partial loadcnt, not a full drain.
// ---------------------------------------------------------------------------

typedef __bf16 bf16_t;
typedef __attribute__((ext_vector_type(16))) __bf16 v16bf;
typedef __attribute__((ext_vector_type(8)))  __bf16 v8bf;
typedef __attribute__((ext_vector_type(8)))  float  v8f;

// ---------------- WMMA fragment loads --------------------------------------
// A (M x K row-major), 16x32 bf16 tile. Layout used here: lanes 0-15 -> M=0..15,
// elements 0..15 = K(16*half .. 16*half+15) (same k order as B).
__device__ inline v16bf load_a_frag(const bf16_t* __restrict__ A, int lda,
                                    int m0, int k0, int lane) {
  const int hh = lane >> 4;
  const int r  = lane & 15;
  const bf16_t* p = A + (size_t)(m0 + r) * lda + k0 + 16 * hh;
  v8bf lo = *(const v8bf*)(p);
  v8bf hi = *(const v8bf*)(p + 8);
  return __builtin_shufflevector(lo, hi, 0, 1, 2, 3, 4, 5, 6, 7,
                                 8, 9, 10, 11, 12, 13, 14, 15);
}

// B (K x N) supplied as BT (N x K row-major). B layout used here (same k order as A) for 32x16 tile:
// lane L: N = L&15, elements e: K = k0 + 16*(L>>4) + e  -> one contiguous
// 32-byte load per lane.
__device__ inline v16bf load_b_frag(const bf16_t* __restrict__ BT, int ldb,
                                    int n0, int k0, int lane) {
  const int hh = lane >> 4;
  const int r  = lane & 15;
  const bf16_t* p = BT + (size_t)(n0 + r) * ldb + k0 + 16 * hh;
  return *(const v16bf*)(p);
}

// Fragment set for one 16x64x32 step (A tile + 4 B tiles).
struct Frags {
  v16bf a, b0, b1, b2, b3;
};
__device__ inline Frags load_frags(const bf16_t* __restrict__ A,
                                   const bf16_t* __restrict__ BT, int K,
                                   int m0, int k, int lane) {
  Frags f;
  f.a  = load_a_frag(A, K, m0, k, lane);
  f.b0 = load_b_frag(BT, K,  0, k, lane);
  f.b1 = load_b_frag(BT, K, 16, k, lane);
  f.b2 = load_b_frag(BT, K, 32, k, lane);
  f.b3 = load_b_frag(BT, K, 48, k, lane);
  return f;
}

// ---------------- Generic wave-tile GEMM -----------------------------------
// Y[M x N](f32, leading dim ldY, column offset colOff) =
//   A[M x K](bf16) @ BT[N x K](bf16)^T  (+ bias[n]) (+= existing Y if accum)
// Each wave computes a 16x64 tile: 4 accumulators, K-loop step 32.
template <int RELU>
__global__ __launch_bounds__(256) void gemm_bf16_kernel(
    const bf16_t* __restrict__ A, const bf16_t* __restrict__ BT,
    const float* __restrict__ bias, float* __restrict__ Y,
    int M, int N, int K, int ldY, int colOff, int accum) {
  const int lane   = threadIdx.x & 31;
  const int nTiles = N >> 6;
  const int mTiles = M >> 4;
  // wave-uniform tile id -> SGPR so loop control / bases stay scalar
  const int tile =
      __builtin_amdgcn_readfirstlane(blockIdx.x * 8 + (threadIdx.x >> 5));
  if (tile >= mTiles * nTiles) return;  // wave-uniform: EXEC stays all-ones
  const int m0 = (tile / nTiles) << 4;
  const int n0 = (tile % nTiles) << 6;

  v8f acc0 = {}, acc1 = {}, acc2 = {}, acc3 = {};
  for (int k0 = 0; k0 < K; k0 += 32) {
    v16bf a  = load_a_frag(A, K, m0, k0, lane);
    v16bf b0 = load_b_frag(BT, K, n0 +  0, k0, lane);
    v16bf b1 = load_b_frag(BT, K, n0 + 16, k0, lane);
    v16bf b2 = load_b_frag(BT, K, n0 + 32, k0, lane);
    v16bf b3 = load_b_frag(BT, K, n0 + 48, k0, lane);
    acc0 = __builtin_amdgcn_wmma_f32_16x16x32_bf16(false, a, false, b0,
                                                   (short)0, acc0, false, false);
    acc1 = __builtin_amdgcn_wmma_f32_16x16x32_bf16(false, a, false, b1,
                                                   (short)0, acc1, false, false);
    acc2 = __builtin_amdgcn_wmma_f32_16x16x32_bf16(false, a, false, b2,
                                                   (short)0, acc2, false, false);
    acc3 = __builtin_amdgcn_wmma_f32_16x16x32_bf16(false, a, false, b3,
                                                   (short)0, acc3, false, false);
  }

  // Epilogue. C/D layout: VGPR v, lane L -> M = m0+v+8*(L>>4), N = n0+16f+(L&15)
  const int hh = lane >> 4;
  const int r  = lane & 15;
#pragma unroll
  for (int f = 0; f < 4; ++f) {
    const int n = n0 + 16 * f + r;
    const float bv = bias ? bias[n] : 0.0f;
    v8f a = (f == 0) ? acc0 : (f == 1) ? acc1 : (f == 2) ? acc2 : acc3;
#pragma unroll
    for (int v = 0; v < 8; ++v) {
      const int m = m0 + v + 8 * hh;
      const size_t off = (size_t)m * ldY + colOff + n;
      float val = a[v] + bv;
      if (accum) val += Y[off];
      if (RELU) val = fmaxf(val, 0.0f);
      Y[off] = val;
    }
  }
}

// ---------------- Split-K GEMM (N == 64) -----------------------------------
// part[p] (M x 64) = A[:, p*K/chunks .. ) @ BT^T chunk. Deterministic: each
// chunk has its own buffer; reduction below sums in fixed order.
// K/chunks must be a multiple of 32.
__global__ __launch_bounds__(256) void gemm_bf16_splitk_kernel(
    const bf16_t* __restrict__ A, const bf16_t* __restrict__ BT,
    float* __restrict__ part, int M, int K, int chunks) {
  const int lane   = threadIdx.x & 31;
  const int mTiles = M >> 4;
  const int tile =
      __builtin_amdgcn_readfirstlane(blockIdx.x * 8 + (threadIdx.x >> 5));
  if (tile >= mTiles * chunks) return;  // wave-uniform
  const int m0   = (tile / chunks) << 4;
  const int p    = tile % chunks;
  const int kLen = K / chunks;
  const int kBeg = p * kLen;
  const int kEnd = kBeg + kLen;

  const int hh = lane >> 4;
  const int r  = lane & 15;
  const bf16_t* aRow = A + (size_t)(m0 + r) * K;

  v8f acc0 = {}, acc1 = {}, acc2 = {}, acc3 = {};

  // Software pipeline: fragments for step k are loaded one iteration ahead,
  // so each WMMA group waits only on the previous iteration's loads while the
  // next iteration's 5 loads stay in flight (partial loadcnt wait).
  Frags cur = load_frags(A, BT, K, m0, kBeg, lane);
  for (int k0 = kBeg; k0 < kEnd - 32; k0 += 32) {
    __builtin_prefetch(aRow + k0 + 512, 0, 3);  // global_prefetch_b8, ~1KB ahead
    Frags nxt = load_frags(A, BT, K, m0, k0 + 32, lane);
    acc0 = __builtin_amdgcn_wmma_f32_16x16x32_bf16(false, cur.a, false, cur.b0,
                                                   (short)0, acc0, false, false);
    acc1 = __builtin_amdgcn_wmma_f32_16x16x32_bf16(false, cur.a, false, cur.b1,
                                                   (short)0, acc1, false, false);
    acc2 = __builtin_amdgcn_wmma_f32_16x16x32_bf16(false, cur.a, false, cur.b2,
                                                   (short)0, acc2, false, false);
    acc3 = __builtin_amdgcn_wmma_f32_16x16x32_bf16(false, cur.a, false, cur.b3,
                                                   (short)0, acc3, false, false);
    cur = nxt;
  }
  // peeled last step
  acc0 = __builtin_amdgcn_wmma_f32_16x16x32_bf16(false, cur.a, false, cur.b0,
                                                 (short)0, acc0, false, false);
  acc1 = __builtin_amdgcn_wmma_f32_16x16x32_bf16(false, cur.a, false, cur.b1,
                                                 (short)0, acc1, false, false);
  acc2 = __builtin_amdgcn_wmma_f32_16x16x32_bf16(false, cur.a, false, cur.b2,
                                                 (short)0, acc2, false, false);
  acc3 = __builtin_amdgcn_wmma_f32_16x16x32_bf16(false, cur.a, false, cur.b3,
                                                 (short)0, acc3, false, false);

  float* out = part + (size_t)p * M * 64;
#pragma unroll
  for (int f = 0; f < 4; ++f) {
    const int n = 16 * f + r;
    v8f a = (f == 0) ? acc0 : (f == 1) ? acc1 : (f == 2) ? acc2 : acc3;
#pragma unroll
    for (int v = 0; v < 8; ++v) {
      const int m = m0 + v + 8 * hh;
      out[(size_t)m * 64 + n] = a[v];
    }
  }
}

// Fixed-order split-K reduction: Y[m][colOff+n] = sum_p part[p][m][n].
__global__ __launch_bounds__(256) void reduce_splitk_kernel(
    const float* __restrict__ part, float* __restrict__ Y, int M, int chunks,
    int ldY, int colOff) {
  const size_t i = (size_t)blockIdx.x * 256 + threadIdx.x;  // over M*64
  if (i >= (size_t)M * 64) return;
  const int m = (int)(i >> 6), n = (int)(i & 63);
  float s = 0.0f;
  for (int p = 0; p < chunks; ++p) s += part[(size_t)p * M * 64 + i];
  Y[(size_t)m * ldY + colOff + n] = s;
}

// ---------------- Conversions / transposes ---------------------------------
__global__ void cvt_f32_bf16_kernel(const float* __restrict__ in,
                                    bf16_t* __restrict__ out, size_t n) {
  size_t i = (size_t)blockIdx.x * blockDim.x + threadIdx.x;
  const size_t stride = (size_t)gridDim.x * blockDim.x;
  for (; i < n; i += stride) out[i] = (bf16_t)in[i];
}

// in: (M x 64) f32 row-major -> out: (64 x M) bf16 row-major (the BT form).
__global__ __launch_bounds__(256) void transpose64_kernel(
    const float* __restrict__ in, bf16_t* __restrict__ out, int M) {
  __shared__ float tile[32][65];
  const int m0 = blockIdx.x * 32;
  const int t  = threadIdx.x;
#pragma unroll
  for (int i = 0; i < 8; ++i) {
    const int idx = t + 256 * i;           // 0..2047 = 32 rows x 64 cols
    const int rr = idx >> 6, cc = idx & 63;
    tile[rr][cc] = in[(size_t)(m0 + rr) * 64 + cc];
  }
  __syncthreads();
#pragma unroll
  for (int i = 0; i < 8; ++i) {
    const int idx = t + 256 * i;
    const int n = idx >> 5, mm = idx & 31;  // n: 0..63, mm: 0..31
    out[(size_t)n * M + m0 + mm] = (bf16_t)tile[mm][n];
  }
}

// General transpose + convert: in (R x C) f32 -> out (C x R) bf16.
__global__ __launch_bounds__(256) void transpose_cvt_kernel(
    const float* __restrict__ in, bf16_t* __restrict__ out, int R, int C) {
  __shared__ float tile[32][33];
  const int r0 = blockIdx.y * 32, c0 = blockIdx.x * 32;
  const int t = threadIdx.x;
#pragma unroll
  for (int i = 0; i < 4; ++i) {
    const int idx = t + 256 * i;            // 0..1023
    const int rr = idx >> 5, cc = idx & 31;
    tile[rr][cc] = in[(size_t)(r0 + rr) * C + c0 + cc];
  }
  __syncthreads();
#pragma unroll
  for (int i = 0; i < 4; ++i) {
    const int idx = t + 256 * i;
    const int cc = idx >> 5, rr = idx & 31;
    out[(size_t)(c0 + cc) * R + r0 + rr] = (bf16_t)tile[rr][cc];
  }
}

// ---------------- LayerNorm-LSTM cell --------------------------------------
__device__ inline float wred32(float v) {
#pragma unroll
  for (int m = 16; m; m >>= 1) v += __shfl_xor(v, m, 32);
  return v;
}
__device__ inline float sigmoidf_(float x) { return 1.0f / (1.0f + __expf(-x)); }

// One wave32 per row; each lane owns 2 of the 64 features.
__global__ __launch_bounds__(256) void lnlstm_kernel(
    const float* __restrict__ pre,                      // M x 256 (i,f,g,o)
    const float* __restrict__ g_gate, const float* __restrict__ b_gate,  // 4x64
    const float* __restrict__ g_c, const float* __restrict__ b_c,        // 64
    float* __restrict__ h, float* __restrict__ c, int M) {
  const int row = blockIdx.x * 8 + (threadIdx.x >> 5);
  if (row >= M) return;
  const int lane = threadIdx.x & 31;
  const int f0 = 2 * lane, f1 = f0 + 1;

  float gate[4][2];
#pragma unroll
  for (int g = 0; g < 4; ++g) {
    const float x0 = pre[(size_t)row * 256 + g * 64 + f0];
    const float x1 = pre[(size_t)row * 256 + g * 64 + f1];
    const float s  = wred32(x0 + x1);
    const float sq = wred32(x0 * x0 + x1 * x1);
    const float mu  = s * (1.0f / 64.0f);
    const float var = sq * (1.0f / 64.0f) - mu * mu;
    const float rs  = rsqrtf(var + 1e-5f);
    gate[g][0] = (x0 - mu) * rs * g_gate[g * 64 + f0] + b_gate[g * 64 + f0];
    gate[g][1] = (x1 - mu) * rs * g_gate[g * 64 + f1] + b_gate[g * 64 + f1];
  }
  const float c0 = c[(size_t)row * 64 + f0];
  const float c1 = c[(size_t)row * 64 + f1];
  const float cn0 = sigmoidf_(gate[1][0]) * c0 + sigmoidf_(gate[0][0]) * tanhf(gate[2][0]);
  const float cn1 = sigmoidf_(gate[1][1]) * c1 + sigmoidf_(gate[0][1]) * tanhf(gate[2][1]);
  const float s  = wred32(cn0 + cn1);
  const float sq = wred32(cn0 * cn0 + cn1 * cn1);
  const float mu  = s * (1.0f / 64.0f);
  const float var = sq * (1.0f / 64.0f) - mu * mu;
  const float rs  = rsqrtf(var + 1e-5f);
  const float ln0 = (cn0 - mu) * rs * g_c[f0] + b_c[f0];
  const float ln1 = (cn1 - mu) * rs * g_c[f1] + b_c[f1];
  h[(size_t)row * 64 + f0] = sigmoidf_(gate[3][0]) * tanhf(ln0);
  h[(size_t)row * 64 + f1] = sigmoidf_(gate[3][1]) * tanhf(ln1);
  c[(size_t)row * 64 + f0] = cn0;
  c[(size_t)row * 64 + f1] = cn1;
}

// ---------------- State init / vote reduction ------------------------------
__global__ void init_state_kernel(float* hV, float* cV, float* hC, float* cC,
                                  int nV, int nC) {
  const size_t i = (size_t)blockIdx.x * 256 + threadIdx.x;
  if (i < (size_t)nV) { hV[i] = 1.0f; cV[i] = 0.0f; }
  if (i < (size_t)nC) { hC[i] = 0.125f; cC[i] = 0.0f; }  // 1/sqrt(64)
}

__global__ __launch_bounds__(256) void vote_kernel(const float* __restrict__ vv,
                                                   float* __restrict__ out,
                                                   int perGraph) {
  __shared__ float s[256];
  const int g = blockIdx.x, t = threadIdx.x;
  const float* p = vv + (size_t)g * perGraph;
  float acc = 0.0f;
  for (int i = t; i < perGraph; i += 256) acc += p[i];
  s[t] = acc;
  __syncthreads();
  for (int k = 128; k; k >>= 1) {
    if (t < k) s[t] += s[t + k];
    __syncthreads();
  }
  if (t == 0) out[g] = 1.0f / (1.0f + __expf(-s[0] / (float)perGraph));
}

// ---------------- Host-side helpers ----------------------------------------
static inline void launch_cvt(const float* in, bf16_t* out, size_t n,
                              hipStream_t s) {
  size_t blocks = (n + 255) / 256;
  if (blocks > 8192) blocks = 8192;
  cvt_f32_bf16_kernel<<<dim3((unsigned)blocks), dim3(256), 0, s>>>(in, out, n);
}

static inline void launch_gemm(const bf16_t* A, const bf16_t* BT,
                               const float* bias, float* Y, int M, int N, int K,
                               int ldY, int colOff, int accum, int relu,
                               hipStream_t s) {
  const int tiles = (M / 16) * (N / 64);
  const int blocks = (tiles + 7) / 8;
  if (relu)
    gemm_bf16_kernel<1><<<dim3(blocks), dim3(256), 0, s>>>(A, BT, bias, Y, M, N,
                                                           K, ldY, colOff, accum);
  else
    gemm_bf16_kernel<0><<<dim3(blocks), dim3(256), 0, s>>>(A, BT, bias, Y, M, N,
                                                           K, ldY, colOff, accum);
}

// Split-K GEMM (N==64) + deterministic reduction into Y[m][colOff + n].
static inline void launch_gemm_splitk(const bf16_t* A, const bf16_t* BT,
                                      float* part, float* Y, int M, int K,
                                      int chunks, int ldY, int colOff,
                                      hipStream_t s) {
  const int tiles = (M / 16) * chunks;
  gemm_bf16_splitk_kernel<<<dim3((tiles + 7) / 8), dim3(256), 0, s>>>(
      A, BT, part, M, K, chunks);
  const int elems = M * 64;
  reduce_splitk_kernel<<<dim3((elems + 255) / 256), dim3(256), 0, s>>>(
      part, Y, M, chunks, ldY, colOff);
}

extern "C" void kernel_launch(void* const* d_in, const int* in_sizes, int n_in,
                              void* d_out, int out_size, void* d_ws,
                              size_t ws_size, hipStream_t stream) {
  (void)in_sizes; (void)n_in; (void)out_size; (void)ws_size;
  const int NV = 8192, NC = 1024, E = 64, NG = 16, TMAX = 32;
  const int SPLIT_VV = 4;   // M_vv @ hV  : 512 -> 2048 waves (kLen 2048)
  const int SPLIT_CT = 16;  // M_vc^T @ V_:  64 -> 1024 waves (kLen 512)

  // ---- inputs (setup_inputs() insertion order, params flattened recursively)
  const float* Mvv_f = (const float*)d_in[0];
  const float* Mvc_f = (const float*)d_in[1];
  const float* mC_W1 = (const float*)d_in[2];  const float* mC_b1 = (const float*)d_in[3];
  const float* mC_W2 = (const float*)d_in[4];  const float* mC_b2 = (const float*)d_in[5];
  const float* mV_W1 = (const float*)d_in[6];  const float* mV_b1 = (const float*)d_in[7];
  const float* mV_W2 = (const float*)d_in[8];  const float* mV_b2 = (const float*)d_in[9];
  const float* vt_W1 = (const float*)d_in[10]; const float* vt_b1 = (const float*)d_in[11];
  const float* vt_W2 = (const float*)d_in[12]; const float* vt_b2 = (const float*)d_in[13];
  const float* lv_Wx = (const float*)d_in[14]; const float* lv_Wh = (const float*)d_in[15];
  const float* lv_b  = (const float*)d_in[16]; const float* lv_gg = (const float*)d_in[17];
  const float* lv_bg = (const float*)d_in[18]; const float* lv_gc = (const float*)d_in[19];
  const float* lv_bc = (const float*)d_in[20];
  const float* lc_Wx = (const float*)d_in[21]; const float* lc_Wh = (const float*)d_in[22];
  const float* lc_b  = (const float*)d_in[23]; const float* lc_gg = (const float*)d_in[24];
  const float* lc_bg = (const float*)d_in[25]; const float* lc_gc = (const float*)d_in[26];
  const float* lc_bc = (const float*)d_in[27];

  // ---- workspace carve (256B aligned slabs)
  char* wp = (char*)d_ws;
  auto take = [&](size_t bytes) -> char* {
    char* r = wp;
    wp += (bytes + 255) & ~(size_t)255;
    return r;
  };
  bf16_t* Mvv_bf  = (bf16_t*)take((size_t)NV * NV * 2);   // 128 MiB, L2-resident
  bf16_t* Mvc_bf  = (bf16_t*)take((size_t)NV * NC * 2);
  bf16_t* MvcT_bf = (bf16_t*)take((size_t)NV * NC * 2);
  bf16_t* WC1 = (bf16_t*)take(64 * 64 * 2);  bf16_t* WC2 = (bf16_t*)take(64 * 64 * 2);
  bf16_t* WV1 = (bf16_t*)take(64 * 64 * 2);  bf16_t* WV2 = (bf16_t*)take(64 * 64 * 2);
  bf16_t* WT1 = (bf16_t*)take(64 * 64 * 2);  bf16_t* WT2 = (bf16_t*)take(64 * 64 * 2);
  bf16_t* WxV = (bf16_t*)take(256 * 128 * 2);
  bf16_t* WhV = (bf16_t*)take(256 * 64 * 2);
  bf16_t* WxC = (bf16_t*)take(256 * 64 * 2);
  bf16_t* WhC = (bf16_t*)take(256 * 64 * 2);
  float*  hV = (float*)take((size_t)NV * E * 4);
  float*  cV = (float*)take((size_t)NV * E * 4);
  float*  hC = (float*)take((size_t)NC * E * 4);
  float*  cC = (float*)take((size_t)NC * E * 4);
  bf16_t* hV_bf = (bf16_t*)take((size_t)NV * E * 2);
  bf16_t* hVT   = (bf16_t*)take((size_t)NV * E * 2);
  bf16_t* hC_bf = (bf16_t*)take((size_t)NC * E * 2);
  float*  t0    = (float*)take((size_t)NC * E * 4);   // mlpC hidden
  bf16_t* t0_bf = (bf16_t*)take((size_t)NC * E * 2);
  float*  t1    = (float*)take((size_t)NC * E * 4);   // mlpC out
  bf16_t* CmT   = (bf16_t*)take((size_t)NC * E * 2);
  float*  t2    = (float*)take((size_t)NV * E * 4);   // mlpV/vote hidden
  bf16_t* t2_bf = (bf16_t*)take((size_t)NV * E * 2);
  float*  t3    = (float*)take((size_t)NV * E * 4);   // mlpV out
  bf16_t* VT    = (bf16_t*)take((size_t)NV * E * 2);
  float*  xv    = (float*)take((size_t)NV * 2 * E * 4);   // concat [Mvv@V | Mvc@Cm]
  bf16_t* xv_bf = (bf16_t*)take((size_t)NV * 2 * E * 2);
  float*  pre_v = (float*)take((size_t)NV * 4 * E * 4);
  float*  xc    = (float*)take((size_t)NC * E * 4);
  bf16_t* xc_bf = (bf16_t*)take((size_t)NC * E * 2);
  float*  pre_c = (float*)take((size_t)NC * 4 * E * 4);
  float*  vv    = (float*)take((size_t)NV * E * 4);
  // split-K partial buffer: max(NV*64*SPLIT_VV, NC*64*SPLIT_CT) floats = 8 MiB
  float*  part  = (float*)take((size_t)NV * 64 * SPLIT_VV * 4);

  // ---- one-time conversions
  launch_cvt(Mvv_f, Mvv_bf, (size_t)NV * NV, stream);
  launch_cvt(Mvc_f, Mvc_bf, (size_t)NV * NC, stream);
  transpose_cvt_kernel<<<dim3(NC / 32, NV / 32), dim3(256), 0, stream>>>(
      Mvc_f, MvcT_bf, NV, NC);
  launch_cvt(mC_W1, WC1, 64 * 64, stream);  launch_cvt(mC_W2, WC2, 64 * 64, stream);
  launch_cvt(mV_W1, WV1, 64 * 64, stream);  launch_cvt(mV_W2, WV2, 64 * 64, stream);
  launch_cvt(vt_W1, WT1, 64 * 64, stream);  launch_cvt(vt_W2, WT2, 64 * 64, stream);
  launch_cvt(lv_Wx, WxV, 256 * 128, stream);
  launch_cvt(lv_Wh, WhV, 256 * 64, stream);
  launch_cvt(lc_Wx, WxC, 256 * 64, stream);
  launch_cvt(lc_Wh, WhC, 256 * 64, stream);
  init_state_kernel<<<dim3((NV * E + 255) / 256), dim3(256), 0, stream>>>(
      hV, cV, hC, cC, NV * E, NC * E);

  // ---- 32 recurrent steps
  for (int t = 0; t < TMAX; ++t) {
    // snapshots of current h states in WMMA operand forms
    launch_cvt(hC, hC_bf, (size_t)NC * E, stream);
    launch_cvt(hV, hV_bf, (size_t)NV * E, stream);
    transpose64_kernel<<<dim3(NV / 32), dim3(256), 0, stream>>>(hV, hVT, NV);

    // Cm = mlpC(hC)
    launch_gemm(hC_bf, WC1, mC_b1, t0, NC, 64, 64, 64, 0, 0, 1, stream);
    launch_cvt(t0, t0_bf, (size_t)NC * E, stream);
    launch_gemm(t0_bf, WC2, mC_b2, t1, NC, 64, 64, 64, 0, 0, 0, stream);
    transpose64_kernel<<<dim3(NC / 32), dim3(256), 0, stream>>>(t1, CmT, NC);

    // V_ = mlpV(hV)
    launch_gemm(hV_bf, WV1, mV_b1, t2, NV, 64, 64, 64, 0, 0, 1, stream);
    launch_cvt(t2, t2_bf, (size_t)NV * E, stream);
    launch_gemm(t2_bf, WV2, mV_b2, t3, NV, 64, 64, 64, 0, 0, 0, stream);
    transpose64_kernel<<<dim3(NV / 32), dim3(256), 0, stream>>>(t3, VT, NV);

    // x_v = [Mvv @ hV | Mvc @ Cm]  -> (NV x 128)
    launch_gemm_splitk(Mvv_bf, hVT, part, xv, NV, NV, SPLIT_VV, 128, 0, stream);
    launch_gemm(Mvc_bf, CmT, nullptr, xv, NV, 64, NC, 128, 64, 0, 0, stream);
    launch_cvt(xv, xv_bf, (size_t)NV * 128, stream);

    // pre_v = x_v @ Wx^T + b ; += hV @ Wh^T
    launch_gemm(xv_bf, WxV, lv_b, pre_v, NV, 256, 128, 256, 0, 0, 0, stream);
    launch_gemm(hV_bf, WhV, nullptr, pre_v, NV, 256, 64, 256, 0, 1, 0, stream);
    lnlstm_kernel<<<dim3(NV / 8), dim3(256), 0, stream>>>(
        pre_v, lv_gg, lv_bg, lv_gc, lv_bc, hV, cV, NV);

    // x_c = Mvc^T @ V_
    launch_gemm_splitk(MvcT_bf, VT, part, xc, NC, NV, SPLIT_CT, 64, 0, stream);
    launch_cvt(xc, xc_bf, (size_t)NC * E, stream);
    launch_gemm(xc_bf, WxC, lc_b, pre_c, NC, 256, 64, 256, 0, 0, 0, stream);
    launch_gemm(hC_bf, WhC, nullptr, pre_c, NC, 256, 64, 256, 0, 1, 0, stream);
    lnlstm_kernel<<<dim3(NC / 8), dim3(256), 0, stream>>>(
        pre_c, lc_gg, lc_bg, lc_gc, lc_bc, hC, cC, NC);
  }

  // ---- vote head
  launch_cvt(hV, hV_bf, (size_t)NV * E, stream);
  launch_gemm(hV_bf, WT1, vt_b1, t2, NV, 64, 64, 64, 0, 0, 1, stream);
  launch_cvt(t2, t2_bf, (size_t)NV * E, stream);
  launch_gemm(t2_bf, WT2, vt_b2, vv, NV, 64, 64, 64, 0, 0, 0, stream);
  vote_kernel<<<dim3(NG), dim3(256), 0, stream>>>(vv, (float*)d_out,
                                                  (NV / NG) * E);

  // outputs: (votes[16], hV[8192*64], hC[1024*64]) concatenated
  hipMemcpyAsync((float*)d_out + NG, hV, (size_t)NV * E * sizeof(float),
                 hipMemcpyDeviceToDevice, stream);
  hipMemcpyAsync((float*)d_out + NG + NV * E, hC, (size_t)NC * E * sizeof(float),
                 hipMemcpyDeviceToDevice, stream);
}
